// TGAT_48704929136975
// MI455X (gfx1250) — compile-verified, matrix operand order checked by reading, not run on hardware
//
#include <hip/hip_runtime.h>
#include <hip/hip_bf16.h>
#include <stdint.h>

#define K_NBR   10
#define EMBED   128
#define TIME_DIM 100
#define N0      3072
#define BSZ     1024
#define M1      (N0 * K_NBR)

typedef __attribute__((ext_vector_type(16))) __bf16 v16bf;
typedef __attribute__((ext_vector_type(8)))  float  v8f;

union Frag {
  v16bf    v;
  uint16_t u[16];
  uint32_t w[8];
};

__device__ __forceinline__ uint16_t f2bf(float f) {
  union { float f; uint32_t u; } x; x.f = f;
  uint32_t r = x.u + 0x7FFFu + ((x.u >> 16) & 1u);   // round-to-nearest-even
  return (uint16_t)(r >> 16);
}
__device__ __forceinline__ float bf2f(uint16_t h) {
  union { uint32_t u; float f; } x; x.u = ((uint32_t)h) << 16;
  return x.f;
}

// A fragment (16x32 bf16 tile, M x K) loaded from an LDS row-major tile.
// Layout per CDNA5 ISA 7.12.2 (16-bit A 16x32): lane%16 = M, lane/16 selects
// K-half offset (+8); regs 0-3 cover K 0..7(+8), regs 4-7 cover K 16..23(+8).
__device__ __forceinline__ v16bf frag_a_lds(const uint16_t* tile, int pitch, int kc, int lane) {
  Frag f;
  int row = lane & 15;
  int hi8 = (lane >> 4) * 8;
  const uint16_t* p = tile + row * pitch + kc * 32 + hi8;
#pragma unroll
  for (int r = 0; r < 8; ++r) {
    int k0 = ((r < 4) ? 0 : 16) + (r & 3) * 2;
    f.w[r] = *(const uint32_t*)(p + k0);          // merges into ds_load_b128
  }
  return f.v;
}

// B fragment (32x16 bf16, K x N), pre-packed in global memory so each lane
// reads 32 contiguous bytes.
__device__ __forceinline__ v16bf frag_b_glb(const uint16_t* wf, int kc, int nt, int lane) {
  Frag f;
  const uint32_t* p = (const uint32_t*)(wf + (((size_t)(kc * 8 + nt) * 32 + lane) << 4));
#pragma unroll
  for (int r = 0; r < 8; ++r) f.w[r] = p[r];
  return f.v;
}

__device__ __forceinline__ v8f wmma_bf16(v16bf a, v16bf b, v8f c) {
  return __builtin_amdgcn_wmma_f32_16x16x32_bf16(false, a, false, b, (short)0, c, false, false);
}

// ---------------------------------------------------------------------------
// Pre-pack a (din x 128) f32 weight matrix into bf16 WMMA B-fragments.
// Grid: (KP/32)*8 blocks of 32 threads.  K rows beyond din are zero-padded.
// ---------------------------------------------------------------------------
__global__ void pack_w(const float* __restrict__ W, int din, uint16_t* __restrict__ out) {
  int kc   = blockIdx.x >> 3;
  int nt   = blockIdx.x & 7;
  int lane = threadIdx.x;
  int hi8  = (lane >> 4) * 8;
  int n    = nt * 16 + (lane & 15);
  uint16_t* o = out + (((size_t)(kc * 8 + nt) * 32 + lane) << 4);
#pragma unroll
  for (int i = 0; i < 16; ++i) {
    int r = i >> 1, h = i & 1;
    int k = kc * 32 + ((r < 4) ? 0 : 16) + (r & 3) * 2 + h + hi8;
    o[i] = f2bf((k < din) ? W[(size_t)k * EMBED + n] : 0.f);
  }
}

// ---------------------------------------------------------------------------
// Fused temporal attention layer.  One workgroup (256 thr = 8 wave32) handles
// 8 dst rows; each wave owns one 16-col n-tile so Wk/Wv WMMA B-fragments stay
// register-resident across all 5 kv tiles (5x less L2 weight traffic).
// REQUIRES: M % 8 == 0 (true for all launches: 3072, 30720).
// ---------------------------------------------------------------------------
__global__ __launch_bounds__(256) void attn_kernel(
    int M,
    const float* __restrict__ hdst_dense, const int* __restrict__ idx_dst,
    const float* __restrict__ dst_feats,
    const float* __restrict__ hsrc_dense, const int* __restrict__ idx_src,
    const float* __restrict__ src_feats,
    const float* __restrict__ edge_feats, const int* __restrict__ eidx,
    const float* __restrict__ t_dst, const float* __restrict__ t_src,
    const uint16_t* __restrict__ WqF, const uint16_t* __restrict__ WkF,
    const uint16_t* __restrict__ WvF, const uint16_t* __restrict__ WoF,
    const float* __restrict__ bq, const float* __restrict__ bk,
    const float* __restrict__ bv, const float* __restrict__ bo,
    const float* __restrict__ tw, const float* __restrict__ tb,
    float* __restrict__ h_out)
{
  __shared__ uint16_t Aq[16 * 256];     //  8 KB: q-input tile, later [h_dst | o]
  __shared__ uint16_t Akv[16 * 384];    // 12 KB: kv-input tile (reused per tile)
  __shared__ float    qbuf[16 * 128];   //  8 KB: q rows (f32; rows 8..15 scratch)
  __shared__ float    ktmp[16 * 128];   //  8 KB: k tile (f32, reused)
  __shared__ float    sc[8 * 2 * 10];   //  scores / softmax weights
  __shared__ uint16_t vbuf[80 * 128];   // 20 KB: all V rows (bf16)

  const int tid  = threadIdx.x;
  const int lane = tid & 31;
  const int wave = tid >> 5;            // = this wave's n-tile (0..7)
  const int m0   = blockIdx.x * 8;
  (void)M;

  // ---- Stage 1: q input tile = [h_dst(128) | cos(tb)(100) | pad] ----------
  for (int e = tid; e < 16 * 256; e += 256) {
    int row = e >> 8, col = e & 255;
    float val = 0.f;
    if (row < 8) {
      if (col < 128) {
        const float* hp = idx_dst ? (dst_feats + (size_t)idx_dst[m0 + row] * 128)
                                  : (hdst_dense + (size_t)(m0 + row) * 128);
        val = hp[col];
      } else if (col < 128 + TIME_DIM) {
        val = __cosf(tb[col - 128]);
      }
    }
    Aq[e] = f2bf(val);
  }
  __syncthreads();

  // ---- Stage 2: q = q_in @ Wq + bq  (one n-tile per wave) -----------------
  {
    v8f acc = {};
#pragma unroll
    for (int kc = 0; kc < 8; ++kc)
      acc = wmma_bf16(frag_a_lds(Aq, 256, kc, lane), frag_b_glb(WqF, kc, wave, lane), acc);
    int n = wave * 16 + (lane & 15);
    int hi = lane >> 4;
    float bqv = bq[n];
#pragma unroll
    for (int r = 0; r < 8; ++r)
      qbuf[(r + hi * 8) * 128 + n] = acc[r] + bqv;   // rows 8..15 are scratch
  }

  // ---- Preload this wave's Wk/Wv fragments: register-resident across tiles
  v16bf wk_r[12], wv_r[12];
#pragma unroll
  for (int kc = 0; kc < 12; ++kc) {
    wk_r[kc] = frag_b_glb(WkF, kc, wave, lane);
    wv_r[kc] = frag_b_glb(WvF, kc, wave, lane);
  }
  __syncthreads();

  // ---- Stage 3: 5 kv tiles: gather+phit build, K/V GEMMs, scores ----------
  for (int t = 0; t < 5; ++t) {
    {
      int row = tid >> 4;            // 16 rows, 16 threads each
      int c0  = (tid & 15) * 24;
      int j   = t * 16 + row;        // kv row within WG: j = dl*10 + kk
      int dl  = j / 10, kk = j - dl * 10;
      int m   = m0 + dl;
      int s   = m * K_NBR + kk;
      const float* srcp = idx_src ? (src_feats + (size_t)idx_src[s] * 128)
                                  : (hsrc_dense + (size_t)s * 128);
      const float* ep   = edge_feats + (size_t)eidx[s] * 128;
      float dtv = t_dst[m] - t_src[s];
      // Prefetch next tile's gathered rows (global_prefetch_b8) to hide the
      // random-access HBM latency that dominates this kernel.
      if (t < 4) {
        int j2  = j + 16;
        int dl2 = j2 / 10, kk2 = j2 - dl2 * 10;
        int s2  = (m0 + dl2) * K_NBR + kk2;
        const float* sp2 = idx_src ? (src_feats + (size_t)idx_src[s2] * 128)
                                   : (hsrc_dense + (size_t)s2 * 128);
        const float* ep2 = edge_feats + (size_t)eidx[s2] * 128;
        int pb = (tid & 15) * 32;   // 16 threads cover the 512-byte row
        __builtin_prefetch((const char*)sp2 + pb, 0, 3);
        __builtin_prefetch((const char*)ep2 + pb, 0, 3);
      }
      for (int cc = 0; cc < 24; ++cc) {
        int col = c0 + cc;
        float val;
        if (col < 128)       val = srcp[col];
        else if (col < 256)  val = ep[col - 128];
        else if (col < 356) { int ci = col - 256; val = __cosf(dtv * tw[ci] + tb[ci]); }
        else                 val = 0.f;
        Akv[row * 384 + col] = f2bf(val);
      }
    }
    __syncthreads();

    {
      v8f ak = {}, av = {};
#pragma unroll
      for (int kc = 0; kc < 12; ++kc) {
        v16bf a = frag_a_lds(Akv, 384, kc, lane);
        ak = wmma_bf16(a, wk_r[kc], ak);
        av = wmma_bf16(a, wv_r[kc], av);
      }
      int n = wave * 16 + (lane & 15);
      int hi = lane >> 4;
      float bkv = bk[n], bvv = bv[n];
#pragma unroll
      for (int r = 0; r < 8; ++r) {
        int mr = r + hi * 8;
        ktmp[mr * 128 + n] = ak[r] + bkv;
        vbuf[(t * 16 + mr) * 128 + n] = f2bf(av[r] + bvv);
      }
    }
    __syncthreads();

    {   // scores for this tile's 16 kv rows: 8 lanes per (row, head)
      int pair = tid >> 3, l8 = tid & 7;   // 32 pairs = 16 rows x 2 heads
      int jl = pair >> 1, h = pair & 1;
      int j  = t * 16 + jl;
      int dl = j / 10, kk = j - dl * 10;
      float d = 0.f;
      const float* qp = qbuf + dl * 128 + h * 64;
      const float* kp = ktmp + jl * 128 + h * 64;
#pragma unroll
      for (int c = l8; c < 64; c += 8) d += qp[c] * kp[c];
      d += __shfl_xor(d, 1, 32);
      d += __shfl_xor(d, 2, 32);
      d += __shfl_xor(d, 4, 32);
      if (l8 == 0) sc[(dl * 2 + h) * 10 + kk] = d * 0.125f;   // / sqrt(64)
    }
    __syncthreads();
  }

  // ---- Stage 4: softmax over the 10 neighbors, per (dst,head) -------------
  if (tid < 16) {
    float* s = sc + tid * 10;
    float mx = s[0];
#pragma unroll
    for (int i = 1; i < 10; ++i) mx = fmaxf(mx, s[i]);
    float sum = 0.f, ex[10];
#pragma unroll
    for (int i = 0; i < 10; ++i) { ex[i] = __expf(s[i] - mx); sum += ex[i]; }
    float inv = 1.f / sum;
#pragma unroll
    for (int i = 0; i < 10; ++i) s[i] = ex[i] * inv;
  }
  __syncthreads();

  // ---- Stage 5: o = a @ v, packed bf16 into Aq cols 128..255 --------------
  {
    int dl = tid >> 5;
    int c0 = (tid & 31) * 4;
    for (int cc = 0; cc < 4; ++cc) {
      int c = c0 + cc;
      int h = c >> 6;
      const float* a = sc + (dl * 2 + h) * 10;
      float o = 0.f;
#pragma unroll
      for (int kk = 0; kk < 10; ++kk)
        o += a[kk] * bf2f(vbuf[(dl * 10 + kk) * 128 + c]);
      Aq[dl * 256 + 128 + c] = f2bf(o);
    }
  }
  __syncthreads();

  // ---- Stage 6: out = relu(concat(h_dst, o) @ Wo + bo) --------------------
  {
    v8f acc = {};
#pragma unroll
    for (int kc = 0; kc < 8; ++kc)
      acc = wmma_bf16(frag_a_lds(Aq, 256, kc, lane), frag_b_glb(WoF, kc, wave, lane), acc);
    int n = wave * 16 + (lane & 15);
    int hi = lane >> 4;
    float bov = bo[n];
#pragma unroll
    for (int r = 0; r < 8; ++r) {
      int mr = r + hi * 8;
      if (mr < 8)
        h_out[(size_t)(m0 + mr) * 128 + n] = fmaxf(acc[r] + bov, 0.f);
    }
  }
}

// ---------------------------------------------------------------------------
// Edge predictor: out[pass*B + i] = relu(s@Ws + bs + d@Wd + bd) @ Wo + bo.
// ---------------------------------------------------------------------------
__global__ __launch_bounds__(128) void edge_pred_kernel(
    const float* __restrict__ h2,
    const uint16_t* __restrict__ WsF, const uint16_t* __restrict__ WdF,
    const float* __restrict__ bs, const float* __restrict__ bd,
    const float* __restrict__ Wo, const float* __restrict__ bo,
    float* __restrict__ out)
{
  __shared__ uint16_t As[16 * 128];
  __shared__ uint16_t Ad[16 * 128];
  __shared__ float red[16];
  const int tid  = threadIdx.x;
  const int lane = tid & 31;
  const int wave = tid >> 5;
  const int i0   = blockIdx.x * 16;

  for (int e = tid; e < 16 * 128; e += 128)
    As[e] = f2bf(h2[(size_t)(i0 + (e >> 7)) * 128 + (e & 127)]);

  for (int pass = 0; pass < 2; ++pass) {
    __syncthreads();
    int base = (pass == 0) ? BSZ : 2 * BSZ;
    for (int e = tid; e < 16 * 128; e += 128)
      Ad[e] = f2bf(h2[(size_t)(base + i0 + (e >> 7)) * 128 + (e & 127)]);
    if (tid < 16) red[tid] = 0.f;
    __syncthreads();

#pragma unroll
    for (int w2 = 0; w2 < 2; ++w2) {
      int nt = wave + w2 * 4;
      v8f acc = {};
#pragma unroll
      for (int kc = 0; kc < 4; ++kc) {
        acc = wmma_bf16(frag_a_lds(As, 128, kc, lane), frag_b_glb(WsF, kc, nt, lane), acc);
        acc = wmma_bf16(frag_a_lds(Ad, 128, kc, lane), frag_b_glb(WdF, kc, nt, lane), acc);
      }
      int n = nt * 16 + (lane & 15);
      int hi = lane >> 4;
      float bb = bs[n] + bd[n];
      float wo = Wo[n];
#pragma unroll
      for (int r = 0; r < 8; ++r) {
        int mr = r + hi * 8;
        float hv = fmaxf(acc[r] + bb, 0.f);
        atomicAdd(&red[mr], hv * wo);     // ds_add_f32 on LDS
      }
    }
    __syncthreads();
    if (tid < 16) out[pass * BSZ + i0 + tid] = red[tid] + bo[0];
  }
}

// ---------------------------------------------------------------------------
extern "C" void kernel_launch(void* const* d_in, const int* in_sizes, int n_in,
                              void* d_out, int out_size, void* d_ws, size_t ws_size,
                              hipStream_t stream) {
  (void)in_sizes; (void)n_in; (void)out_size; (void)ws_size;

  const float* node_feats = (const float*)d_in[0];
  const float* edge_feats = (const float*)d_in[1];
  const float* t_root     = (const float*)d_in[2];
  const float* t_nbr1     = (const float*)d_in[3];
  const float* t_nbr2     = (const float*)d_in[4];
  const int*   roots      = (const int*)d_in[5];
  const int*   nbr1       = (const int*)d_in[6];
  const int*   eid1       = (const int*)d_in[7];
  const int*   nbr2       = (const int*)d_in[8];
  const int*   eid2       = (const int*)d_in[9];
  // params pytree flattened: 'ep' (Wd,Wo,Ws,bd,bo,bs) then 'layers'[0..1]
  // each (Wk,Wo,Wq,Wv,bk,bo,bq,bv,tb,tw) — dict keys sorted.
  const float* ep_Wd = (const float*)d_in[10];
  const float* ep_Wo = (const float*)d_in[11];
  const float* ep_Ws = (const float*)d_in[12];
  const float* ep_bd = (const float*)d_in[13];
  const float* ep_bo = (const float*)d_in[14];
  const float* ep_bs = (const float*)d_in[15];
  const float* L_Wk[2] = {(const float*)d_in[16], (const float*)d_in[26]};
  const float* L_Wo[2] = {(const float*)d_in[17], (const float*)d_in[27]};
  const float* L_Wq[2] = {(const float*)d_in[18], (const float*)d_in[28]};
  const float* L_Wv[2] = {(const float*)d_in[19], (const float*)d_in[29]};
  const float* L_bk[2] = {(const float*)d_in[20], (const float*)d_in[30]};
  const float* L_bo[2] = {(const float*)d_in[21], (const float*)d_in[31]};
  const float* L_bq[2] = {(const float*)d_in[22], (const float*)d_in[32]};
  const float* L_bv[2] = {(const float*)d_in[23], (const float*)d_in[33]};
  const float* L_tb[2] = {(const float*)d_in[24], (const float*)d_in[34]};
  const float* L_tw[2] = {(const float*)d_in[25], (const float*)d_in[35]};

  uint8_t* ws = (uint8_t*)d_ws;
  size_t off = 0;
  auto alloc = [&](size_t bytes) -> uint8_t* {
    uint8_t* p = ws + off;
    off += (bytes + 255) & ~(size_t)255;
    return p;
  };
  uint16_t *WqF[2], *WkF[2], *WvF[2], *WoF[2];
  for (int l = 0; l < 2; ++l) {
    WqF[l] = (uint16_t*)alloc((size_t)256 * 128 * 2);
    WkF[l] = (uint16_t*)alloc((size_t)384 * 128 * 2);
    WvF[l] = (uint16_t*)alloc((size_t)384 * 128 * 2);
    WoF[l] = (uint16_t*)alloc((size_t)256 * 128 * 2);
  }
  uint16_t* WsF = (uint16_t*)alloc((size_t)128 * 128 * 2);
  uint16_t* WdF = (uint16_t*)alloc((size_t)128 * 128 * 2);
  float* h1_root = (float*)alloc((size_t)N0 * 128 * 4);
  float* h1_n    = (float*)alloc((size_t)M1 * 128 * 4);
  float* h2      = (float*)alloc((size_t)N0 * 128 * 4);

  // Pack weights into WMMA B-fragment order (bf16, K zero-padded).
  for (int l = 0; l < 2; ++l) {
    pack_w<<<dim3(8 * 8),  dim3(32), 0, stream>>>(L_Wq[l], 228, WqF[l]);
    pack_w<<<dim3(12 * 8), dim3(32), 0, stream>>>(L_Wk[l], 356, WkF[l]);
    pack_w<<<dim3(12 * 8), dim3(32), 0, stream>>>(L_Wv[l], 356, WvF[l]);
    pack_w<<<dim3(8 * 8),  dim3(32), 0, stream>>>(L_Wo[l], 256, WoF[l]);
  }
  pack_w<<<dim3(4 * 8), dim3(32), 0, stream>>>(ep_Ws, 128, WsF);
  pack_w<<<dim3(4 * 8), dim3(32), 0, stream>>>(ep_Wd, 128, WdF);

  // Layer 1a: roots attending over nbr1.
  attn_kernel<<<N0 / 8, 256, 0, stream>>>(
      N0, nullptr, roots, node_feats, nullptr, nbr1, node_feats,
      edge_feats, eid1, t_root, t_nbr1,
      WqF[0], WkF[0], WvF[0], WoF[0],
      L_bq[0], L_bk[0], L_bv[0], L_bo[0], L_tw[0], L_tb[0], h1_root);
  // Layer 1b: nbr1 rows attending over nbr2.
  attn_kernel<<<M1 / 8, 256, 0, stream>>>(
      M1, nullptr, nbr1, node_feats, nullptr, nbr2, node_feats,
      edge_feats, eid2, t_nbr1, t_nbr2,
      WqF[0], WkF[0], WvF[0], WoF[0],
      L_bq[0], L_bk[0], L_bv[0], L_bo[0], L_tw[0], L_tb[0], h1_n);
  // Layer 2: h1_root attending over h1_n.
  attn_kernel<<<N0 / 8, 256, 0, stream>>>(
      N0, h1_root, nullptr, nullptr, h1_n, nullptr, nullptr,
      edge_feats, eid1, t_root, t_nbr1,
      WqF[1], WkF[1], WvF[1], WoF[1],
      L_bq[1], L_bk[1], L_bv[1], L_bo[1], L_tw[1], L_tb[1], h2);
  // Edge prediction (pos + neg).
  edge_pred_kernel<<<BSZ / 16, 128, 0, stream>>>(
      h2, WsF, WdF, ep_bs, ep_bd, ep_Wo, ep_bo, (float*)d_out);
}
